// DeformableTransformerDecoderLayer_11579231830642
// MI455X (gfx1250) — matrix operand order repaired and checked by reading, not compiled, not run on hardware
//
#include <hip/hip_runtime.h>
#include <hip/hip_bf16.h>

// ---------------- constants (match reference) ----------------
constexpr int kC   = 256;
constexpr int kDFF = 1024;
constexpr int kNH  = 8;
constexpr int kNL  = 4;
constexpr int kNP  = 4;
constexpr int kQ   = 300;
constexpr int kB   = 8;
constexpr int kS   = 16384 + 4096 + 1024 + 256; // 21760
constexpr int kDH  = kC / kNH;                  // 32

// gfx1250 async global->LDS staging (GLOBAL_LOAD_ASYNC_TO_LDS_B128, ASYNCcnt).
// Set to 0 to fall back to VGPR staging if the assembler rejects the mnemonic.
#define GFX1250_ASYNC_LDS 1

#define DEV __device__ __forceinline__

typedef __attribute__((ext_vector_type(16))) _Float16 v16h;
typedef __attribute__((ext_vector_type(8)))  _Float16 v8h;
typedef __attribute__((ext_vector_type(8)))  float    v8f;

union V16u { v16h v; v8h h[2]; _Float16 e[16]; };

DEV v8f wmma32(v16h a, v16h b, v8f c) {
  // D(16x16,f32) = A(16x32,f16) x B(32x16,f16) + C
  return __builtin_amdgcn_wmma_f32_16x16x32_f16(false, a, false, b, (short)0, c,
                                                false, false);
}

#if GFX1250_ASYNC_LDS
// Per-lane 16B DMA: LDS[vdst_off] = MEM[vaddr]  (tracked by ASYNCcnt).
// LDS offset = low 32 bits of the generic pointer (flat LDS aperture drops the
// upper bits, ISA 10.2).
DEV void async_copy16(void* lds_dst, const void* gsrc) {
  const unsigned loff = (unsigned)(unsigned long long)lds_dst;
  asm volatile("global_load_async_to_lds_b128 %0, %1, off"
               :: "v"(loff), "v"(gsrc) : "memory");
}
DEV void wait_async() { asm volatile("s_wait_asynccnt 0x0" ::: "memory"); }
#endif

// ---- fragment builders -------------------------------------------------
// A fragment from an f16 LDS tile stored [m][k]:
// lane l: M = l%16; halves 0..15 -> K = k0..k0+15,
// k0 = 16*(l/16)
DEV v16h frag_a_lds(const _Float16* base, int lda, int lane) {
  const int m  = lane & 15;
  const int k0 = (lane >> 4) << 4;
  const _Float16* p = base + (size_t)m * lda + k0;
  V16u r;
  r.h[0] = *(const v8h*)(p);
  r.h[1] = *(const v8h*)(p + 8);
  return r.v;
}

// A fragment built from f32 LDS tile(s) with on-the-fly f16 conversion
// (optionally fusing A1+A2).
DEV v16h frag_a_f32(const float* p1, const float* p2, int lda, int lane) {
  const int m  = lane & 15;
  const int k0 = (lane >> 4) << 4;
  const float* a = p1 + (size_t)m * lda + k0;
  const float* b = p2 ? (p2 + (size_t)m * lda + k0) : nullptr;
  V16u r;
#pragma unroll
  for (int e = 0; e < 8; ++e) {
    float v0 = a[e], v1 = a[8 + e];
    if (b) { v0 += b[e]; v1 += b[8 + e]; }
    r.e[e]     = (_Float16)v0;
    r.e[8 + e] = (_Float16)v1;
  }
  return r.v;
}

// B fragment from f32 LDS tile stored [n][k]: lane l: N = l%16, K = e + 16*(l/16)
DEV v16h frag_b_f32(const float* p, int ldb, int lane) {
  const int n  = lane & 15;
  const int k0 = (lane >> 4) << 4;
  const float* q = p + (size_t)n * ldb + k0;
  V16u r;
#pragma unroll
  for (int e = 0; e < 16; ++e) r.e[e] = (_Float16)q[e];
  return r.v;
}

// ---------------- generic WMMA GEMM ----------------
// out[M][N] = (A1 + A2?)[M][K] @ W[N][K]^T + bias[N], optional relu, optional row mask->0
// BM=32, BN=64, 128 threads (4 waves), each wave: 2 WMMA accumulators (32x16).
// Tiles staged in f32 via async global->LDS DMA; f16 conversion at fragment build.
__global__ __launch_bounds__(128)
void gemm_kernel(const float* __restrict__ A1, const float* __restrict__ A2,
                 const float* __restrict__ W,  const float* __restrict__ bias,
                 const unsigned char* __restrict__ rmask,
                 float* __restrict__ out, int M, int N, int K, int relu)
{
  __shared__ __attribute__((aligned(16))) float As1[32][32];
  __shared__ __attribute__((aligned(16))) float As2[32][32];
  __shared__ __attribute__((aligned(16))) float BsF[64][32];
  const int tid  = threadIdx.x;
  const int wave = tid >> 5;
  const int lane = tid & 31;
  const int m0 = blockIdx.y * 32;
  const int n0 = blockIdx.x * 64;

  v8f c0 = {}, c1 = {};

  for (int kk = 0; kk < K; kk += 32) {
#if GFX1250_ASYNC_LDS
    // A tiles: 32x32 f32 = 4KB -> 2 issues x 128 lanes x 16B
#pragma unroll
    for (int i = 0; i < 2; ++i) {
      const int idx = tid + i * 128;
      const int row = idx >> 3;
      const int c4  = (idx & 7) << 2;
      async_copy16(&As1[row][c4], A1 + (size_t)(m0 + row) * K + kk + c4);
      if (A2)
        async_copy16(&As2[row][c4], A2 + (size_t)(m0 + row) * K + kk + c4);
    }
    // B tile: 64x32 f32 = 8KB -> 4 issues
#pragma unroll
    for (int i = 0; i < 4; ++i) {
      const int idx = tid + i * 128;
      const int row = idx >> 3;
      const int c4  = (idx & 7) << 2;
      async_copy16(&BsF[row][c4], W + (size_t)(n0 + row) * K + kk + c4);
    }
    wait_async();
#else
#pragma unroll
    for (int i = 0; i < 2; ++i) {
      const int idx = tid + i * 128;
      const int row = idx >> 3;
      const int c4  = (idx & 7) << 2;
      *(float4*)&As1[row][c4] =
          *(const float4*)(A1 + (size_t)(m0 + row) * K + kk + c4);
      if (A2)
        *(float4*)&As2[row][c4] =
            *(const float4*)(A2 + (size_t)(m0 + row) * K + kk + c4);
    }
#pragma unroll
    for (int i = 0; i < 4; ++i) {
      const int idx = tid + i * 128;
      const int row = idx >> 3;
      const int c4  = (idx & 7) << 2;
      *(float4*)&BsF[row][c4] =
          *(const float4*)(W + (size_t)(n0 + row) * K + kk + c4);
    }
#endif
    __syncthreads();

    const float* a2b = A2 ? &As2[0][0] : nullptr;
    const v16h bf = frag_b_f32(&BsF[wave * 16][0], 32, lane);
    const v16h a0 = frag_a_f32(&As1[0][0],  a2b, 32, lane);
    const v16h a1 = frag_a_f32(&As1[16][0], a2b ? a2b + 16 * 32 : nullptr, 32, lane);
    c0 = wmma32(a0, bf, c0);
    c1 = wmma32(a1, bf, c1);
    __syncthreads();
  }

  const int n  = n0 + wave * 16 + (lane & 15);
  const float bn = bias ? bias[n] : 0.f;
  const int mb = (lane >> 4) * 8;
#pragma unroll
  for (int r = 0; r < 8; ++r) {
    int m = m0 + mb + r;
    float v = c0[r] + bn;
    if (relu) v = fmaxf(v, 0.f);
    if (rmask && rmask[m]) v = 0.f;
    out[(size_t)m * N + n] = v;
    m += 16;
    float v2 = c1[r] + bn;
    if (relu) v2 = fmaxf(v2, 0.f);
    if (rmask && rmask[m]) v2 = 0.f;
    out[(size_t)m * N + n] = v2;
  }
}

// ---------------- self attention (per (b,h,16-query tile), 1 wave) ----------------
__global__ __launch_bounds__(32)
void self_attn_kernel(const float* __restrict__ qh, const float* __restrict__ kh,
                      const float* __restrict__ vh, float* __restrict__ sa)
{
  __shared__ __attribute__((aligned(16))) float    lg[16][320];
  __shared__ __attribute__((aligned(16))) _Float16 pr[16][320];
  const int lane = threadIdx.x;
  const int mt = blockIdx.x, h = blockIdx.y, b = blockIdx.z;
  const int mloc = lane & 15, grp = lane >> 4;

  // A fragment: 16 query rows x 32 dims
  V16u aq;
  {
    const int m  = mt * 16 + mloc;
    const int k0 = grp * 16;
#pragma unroll
    for (int e = 0; e < 16; ++e) {
      const int k = k0 + e;
      aq.e[e] = (m < kQ) ? (_Float16)qh[((size_t)(b * kQ + m)) * kC + h * kDH + k]
                         : (_Float16)0.f;
    }
  }
  // logits = q @ k^T  (19 key tiles, K=32 => one WMMA each)
  for (int nt = 0; nt < 19; ++nt) {
    V16u bkf;
    const int n  = nt * 16 + mloc;
    const int kb = grp * 16;
#pragma unroll
    for (int e = 0; e < 16; ++e) {
      bkf.e[e] = (n < kQ) ? (_Float16)kh[((size_t)(b * kQ + n)) * kC + h * kDH + kb + e]
                          : (_Float16)0.f;
    }
    v8f cc = {};
    cc = wmma32(aq.v, bkf.v, cc);
#pragma unroll
    for (int r = 0; r < 8; ++r) lg[grp * 8 + r][nt * 16 + mloc] = cc[r];
  }
  __syncthreads();

  // row softmax (lane = row)
  if (lane < 16) {
    const float sc = 0.17677669529663687f; // 1/sqrt(32)
    float mx = -1e30f;
    for (int k = 0; k < kQ; ++k) { const float v = lg[lane][k] * sc; lg[lane][k] = v; mx = fmaxf(mx, v); }
    float s = 0.f;
    for (int k = 0; k < kQ; ++k) { const float e = expf(lg[lane][k] - mx); s += e; lg[lane][k] = e; }
    const float inv = 1.f / s;
    for (int k = 0; k < 320; ++k)
      pr[lane][k] = (_Float16)((k < kQ) ? lg[lane][k] * inv : 0.f);
  }
  __syncthreads();

  // out = probs(16x320,f16) @ v(320x32), K padded with zero probs
  v8f o0 = {}, o1 = {};
  for (int kt = 0; kt < 10; ++kt) {
    const v16h ap = frag_a_lds(&pr[0][kt * 32], 320, lane);
    V16u bv0, bv1;
    const int kbase = kt * 32 + grp * 16;
#pragma unroll
    for (int e = 0; e < 16; ++e) {
      const int key = kbase + e;
      const bool ok = key < kQ;
      const size_t base = ((size_t)(b * kQ + (ok ? key : 0))) * kC + h * kDH;
      bv0.e[e] = ok ? (_Float16)vh[base + mloc]      : (_Float16)0.f;
      bv1.e[e] = ok ? (_Float16)vh[base + 16 + mloc] : (_Float16)0.f;
    }
    o0 = wmma32(ap, bv0.v, o0);
    o1 = wmma32(ap, bv1.v, o1);
  }
#pragma unroll
  for (int r = 0; r < 8; ++r) {
    const int m = mt * 16 + grp * 8 + r;
    if (m < kQ) {
      sa[((size_t)(b * kQ + m)) * kC + h * kDH + mloc]      = o0[r];
      sa[((size_t)(b * kQ + m)) * kC + h * kDH + 16 + mloc] = o1[r];
    }
  }
}

// ---------------- residual add + LayerNorm (row = (b,q), 256 threads) ----------------
__global__ __launch_bounds__(256)
void add_ln_kernel(const float* __restrict__ x, const float* __restrict__ y,
                   const float* __restrict__ g, const float* __restrict__ bta,
                   float* __restrict__ out)
{
  __shared__ float red[8];
  const int row = blockIdx.x, t = threadIdx.x;
  const size_t base = (size_t)row * kC;
  const float v = x[base + t] + y[base + t];

  float s = v;
  for (int o = 16; o > 0; o >>= 1) s += __shfl_down(s, o, 32);
  if ((t & 31) == 0) red[t >> 5] = s;
  __syncthreads();
  float mean = 0.f;
  for (int i = 0; i < 8; ++i) mean += red[i];
  mean *= (1.f / kC);
  __syncthreads();

  const float d = v - mean;
  float s2 = d * d;
  for (int o = 16; o > 0; o >>= 1) s2 += __shfl_down(s2, o, 32);
  if ((t & 31) == 0) red[t >> 5] = s2;
  __syncthreads();
  float var = 0.f;
  for (int i = 0; i < 8; ++i) var += red[i];
  var *= (1.f / kC);

  out[base + t] = d * rsqrtf(var + 1e-5f) * g[t] + bta[t];
}

// ---------------- attention-weight softmax over NL*NP=16, per (b,q,h) ----------------
__global__ void awsoftmax_kernel(float* __restrict__ aw)
{
  const int i = blockIdx.x * blockDim.x + threadIdx.x;
  if (i >= kB * kQ * kNH) return;
  float* p = aw + (size_t)(i / kNH) * (kNH * 16) + (i % kNH) * 16;
  float mx = -1e30f;
#pragma unroll
  for (int j = 0; j < 16; ++j) mx = fmaxf(mx, p[j]);
  float e[16], s = 0.f;
#pragma unroll
  for (int j = 0; j < 16; ++j) { e[j] = expf(p[j] - mx); s += e[j]; }
  const float inv = 1.f / s;
#pragma unroll
  for (int j = 0; j < 16; ++j) p[j] = e[j] * inv;
}

// ---------------- deformable bilinear sampling: 1 wave per (b,q,h), lane = channel ----------------
__global__ __launch_bounds__(256)
void deform_kernel(const float* __restrict__ value, const float* __restrict__ refp,
                   const float* __restrict__ off, const float* __restrict__ aw,
                   const int* __restrict__ shapes, const int* __restrict__ lstart,
                   float* __restrict__ acc)
{
  const int gw   = (blockIdx.x * blockDim.x + threadIdx.x) >> 5;
  const int lane = threadIdx.x & 31;
  if (gw >= kB * kQ * kNH) return;
  const int h = gw % kNH;
  const int q = (gw / kNH) % kQ;
  const int b = gw / (kNH * kQ);
  const int d = lane; // DH == wave32

  float a = 0.f;
  for (int l = 0; l < kNL; ++l) {
    const int H  = shapes[l * 2 + 0];
    const int W  = shapes[l * 2 + 1];
    const int st = lstart[l];
    const float rx = refp[(((size_t)b * kQ + q) * kNL + l) * 2 + 0];
    const float ry = refp[(((size_t)b * kQ + q) * kNL + l) * 2 + 1];
    for (int p = 0; p < kNP; ++p) {
      const size_t ob = ((size_t)(b * kQ + q)) * (kNH * kNL * kNP * 2) +
                        (size_t)h * (kNL * kNP * 2) + l * (kNP * 2) + p * 2;
      const float lx = rx + off[ob + 0] / (float)W;
      const float ly = ry + off[ob + 1] / (float)H;
      const float x  = lx * W - 0.5f;
      const float y  = ly * H - 0.5f;
      const float x0 = floorf(x), y0 = floorf(y);
      const float wx = x - x0, wy = y - y0;
      const float w = aw[((size_t)(b * kQ + q)) * (kNH * kNL * kNP) +
                         (size_t)h * (kNL * kNP) + l * kNP + p];
#pragma unroll
      for (int dy = 0; dy < 2; ++dy)
#pragma unroll
        for (int dx = 0; dx < 2; ++dx) {
          const float xi = x0 + dx, yi = y0 + dy;
          if (xi >= 0.f && xi < (float)W && yi >= 0.f && yi < (float)H) {
            const float wgt = (dx ? wx : 1.f - wx) * (dy ? wy : 1.f - wy);
            const int idx = (int)yi * W + (int)xi;
            a += w * wgt *
                 value[((size_t)b * kS + st + idx) * kC + h * kDH + d];
          }
        }
    }
  }
  acc[((size_t)(b * kQ + q)) * kC + h * kDH + d] = a;
}

// ---------------- launcher ----------------
extern "C" void kernel_launch(void* const* d_in, const int* in_sizes, int n_in,
                              void* d_out, int out_size, void* d_ws, size_t ws_size,
                              hipStream_t stream)
{
  const float* tgt    = (const float*)d_in[0];
  // d_in[1] = tgt_box (unused by reference)
  const float* qpos   = (const float*)d_in[2];
  const float* refp   = (const float*)d_in[3];
  const float* src    = (const float*)d_in[4];
  const int*   shapes = (const int*)d_in[5];
  const int*   lstart = (const int*)d_in[6];
  const unsigned char* pmask = (const unsigned char*)d_in[7];
  const float* wq = (const float*)d_in[8],  *bq = (const float*)d_in[9];
  const float* wk = (const float*)d_in[10], *bk = (const float*)d_in[11];
  const float* wv = (const float*)d_in[12], *bv = (const float*)d_in[13];
  const float* wo = (const float*)d_in[14], *bo = (const float*)d_in[15];
  const float* w_off  = (const float*)d_in[16], *b_off  = (const float*)d_in[17];
  const float* w_attn = (const float*)d_in[18], *b_attn = (const float*)d_in[19];
  const float* w_val  = (const float*)d_in[20], *b_val  = (const float*)d_in[21];
  const float* w_out  = (const float*)d_in[22], *b_out  = (const float*)d_in[23];
  const float* w1 = (const float*)d_in[24], *b1 = (const float*)d_in[25];
  const float* w2 = (const float*)d_in[26], *b2 = (const float*)d_in[27];
  const float* ln2g = (const float*)d_in[28], *ln2b = (const float*)d_in[29];
  const float* ln1g = (const float*)d_in[30], *ln1b = (const float*)d_in[31];
  const float* ln3g = (const float*)d_in[32], *ln3b = (const float*)d_in[33];

  float* ws = (float*)d_ws;
  size_t cur = 0;
  auto alloc = [&](size_t n) { float* p = ws + cur; cur += n; return p; };
  const size_t MQ = (size_t)kB * kQ;  // 2400
  const size_t MS = (size_t)kB * kS;  // 174080

  float* qh    = alloc(MQ * kC);
  float* khb   = alloc(MQ * kC);
  float* vhb   = alloc(MQ * kC);
  float* sab   = alloc(MQ * kC);
  float* sap   = alloc(MQ * kC);
  float* t_ln2 = alloc(MQ * kC);
  float* offb  = alloc(MQ * (kNH * kNL * kNP * 2));
  float* awb   = alloc(MQ * (kNH * kNL * kNP));
  float* accb  = alloc(MQ * kC);
  float* t2p   = alloc(MQ * kC);
  float* t_ln1 = alloc(MQ * kC);
  float* ffh   = alloc(MQ * kDFF);
  float* ff2   = alloc(MQ * kC);
  float* val   = alloc(MS * kC);  // 178 MB value buffer

  const dim3 blk(128);
  auto gemm = [&](const float* A1, const float* A2, const float* W, const float* bias,
                  const unsigned char* rm, float* out, int M, int N, int K, int relu) {
    dim3 grid(N / 64, M / 32);
    gemm_kernel<<<grid, blk, 0, stream>>>(A1, A2, W, bias, rm, out, M, N, K, relu);
  };

  // --- self attention block ---
  gemm(tgt, qpos, wq, bq, nullptr, qh,  (int)MQ, kC, kC, 0);  // q = tgt+pos @ wq
  gemm(tgt, qpos, wk, bk, nullptr, khb, (int)MQ, kC, kC, 0);
  gemm(tgt, nullptr, wv, bv, nullptr, vhb, (int)MQ, kC, kC, 0);
  self_attn_kernel<<<dim3(19, kNH, kB), 32, 0, stream>>>(qh, khb, vhb, sab);
  gemm(sab, nullptr, wo, bo, nullptr, sap, (int)MQ, kC, kC, 0);
  add_ln_kernel<<<(int)MQ, 256, 0, stream>>>(tgt, sap, ln2g, ln2b, t_ln2);

  // --- deformable attention block ---
  gemm(src, nullptr, w_val, b_val, pmask, val, (int)MS, kC, kC, 0);   // FLOP-dominant GEMM
  gemm(t_ln2, qpos, w_off,  b_off,  nullptr, offb, (int)MQ, kNH * kNL * kNP * 2, kC, 0);
  gemm(t_ln2, qpos, w_attn, b_attn, nullptr, awb,  (int)MQ, kNH * kNL * kNP,     kC, 0);
  awsoftmax_kernel<<<(kB * kQ * kNH + 255) / 256, 256, 0, stream>>>(awb);
  deform_kernel<<<(kB * kQ * kNH * 32) / 256, 256, 0, stream>>>(val, refp, offb, awb,
                                                                shapes, lstart, accb);
  gemm(accb, nullptr, w_out, b_out, nullptr, t2p, (int)MQ, kC, kC, 0);
  add_ln_kernel<<<(int)MQ, 256, 0, stream>>>(t_ln2, t2p, ln1g, ln1b, t_ln1);

  // --- FFN ---
  gemm(t_ln1, nullptr, w1, b1, nullptr, ffh, (int)MQ, kDFF, kC, 1);   // relu
  gemm(ffh,   nullptr, w2, b2, nullptr, ff2, (int)MQ, kC, kDFF, 0);
  add_ln_kernel<<<(int)MQ, 256, 0, stream>>>(t_ln1, ff2, ln3g, ln3b, (float*)d_out);
}
